// GQA_9113920602192
// MI455X (gfx1250) — matrix operand order repaired and checked by reading, not compiled, not run on hardware
//
#include <hip/hip_runtime.h>

// ---------------------------------------------------------------------------
// GQA attention block for MI455X (gfx1250, wave32, WMMA + TDM).
// Pipeline: f32->f16 cvt -> WMMA GEMMs (Q,K,V; A-tile staged via
//           tensor_load_to_lds with LDS padding) -> RoPE repack -> V transpose
//           -> flash attention (online softmax, v_wmma_f32_16x16x32_f16)
//           -> WMMA GEMM output projection (f32 out).
// ---------------------------------------------------------------------------

typedef _Float16 v8h  __attribute__((ext_vector_type(8)));
typedef _Float16 v16h __attribute__((ext_vector_type(16)));
typedef float    v8f  __attribute__((ext_vector_type(8)));
typedef unsigned int v4u __attribute__((ext_vector_type(4)));
typedef int      v8i  __attribute__((ext_vector_type(8)));
typedef int      v4i  __attribute__((ext_vector_type(4)));

constexpr int S  = 2048;
constexpr int D  = 2048;
constexpr int HQ = 32;
constexpr int HKV = 8;
constexpr int HD = 64;
constexpr int G  = 4;

#if defined(__has_builtin)
#if __has_builtin(__builtin_amdgcn_tensor_load_to_lds) && \
    __has_builtin(__builtin_amdgcn_s_wait_tensorcnt)
#define HAVE_TDM 1
#endif
#endif
#ifndef HAVE_TDM
#define HAVE_TDM 0
#endif

__device__ __forceinline__ v16h cat16(v8h lo, v8h hi) {
  return __builtin_shufflevector(lo, hi, 0,1,2,3,4,5,6,7,8,9,10,11,12,13,14,15);
}
// A-fragment (16x32 f16, M x K): lane holds row m=lane%16; lanes<16 hold
// K 0..15, lanes>=16 hold K 16..31 (koff=0 / 8 -> start 2*koff).
__device__ __forceinline__ v16h load_afrag(const _Float16* row, int koff) {
  v8h lo = *(const v8h*)(row + 2 * koff);
  v8h hi = *(const v8h*)(row + 2 * koff + 8);
  return cat16(lo, hi);
}
// B-fragment (32x16 f16, K x N): lane holds column n=lane%16 with 16
// contiguous K values starting at (lane/16)*16 -> one 32B contiguous read.
__device__ __forceinline__ v16h load_bfrag(const _Float16* p) {
  v8h lo = *(const v8h*)(p);
  v8h hi = *(const v8h*)(p + 8);
  return cat16(lo, hi);
}
__device__ __forceinline__ v8f wmma32(v16h a, v16h b, v8f c) {
  return __builtin_amdgcn_wmma_f32_16x16x32_f16(false, a, false, b, (short)0, c,
                                                false, false);
}

#if HAVE_TDM
// TDM 2D tile load: global (row-major, element=2B) -> LDS, with the D# pad
// feature adding 4 DWORDs (16B) after every 16 DWORDs (64B) so a 32-half tile
// row lands on an 80B (= 40-half) padded LDS stride.
// D# packing per CDNA5 ISA 8.3/8.4 (group0: count/lds/global/type,
// group1: data_size, pad, dims, tile dims, stride). Groups 2/3 zero (2D).
// 6-arg builtin form: (v4u g0, v8i g1, v4i g2, v4i g3, v8i extra, i32 cpol).
__device__ __forceinline__ void tdm_load_tile_2d(unsigned lds_byte_addr,
                                                 const void* gaddr,
                                                 unsigned dim0, unsigned dim1,
                                                 unsigned stride0,
                                                 unsigned tile0, unsigned tile1) {
  unsigned long long ga = (unsigned long long)(uintptr_t)gaddr;
  v4u g0;
  g0[0] = 1u;                                   // count=1 (valid), user mode
  g0[1] = lds_byte_addr;                        // lds_addr
  g0[2] = (unsigned)(ga & 0xFFFFFFFFu);         // global_addr[31:0]
  g0[3] = (unsigned)((ga >> 32) & 0x01FFFFFFu)  // global_addr[56:32]
        | (2u << 30);                           // type=2 ("image")
  v8i g1;
  g1[0] = (1 << 16)        // data_size: 1 -> 2 bytes
        | (1 << 20)        // pad_enable
        | (3 << 22)        // pad_interval: code 3 -> every 16 DWORDs
        | (3 << 25);       // pad_amount:   code 3 -> 4 DWORDs
  g1[1] = (int)((dim0 & 0xFFFFu) << 16);                    // tensor_dim0 lo
  g1[2] = (int)((dim0 >> 16) | ((dim1 & 0xFFFFu) << 16));   // dim0 hi|dim1 lo
  g1[3] = (int)((dim1 >> 16) | (tile0 << 16));              // dim1 hi|tile0
  g1[4] = (int)tile1;                                       // tile1, tile2=0
  g1[5] = (int)stride0;                                     // dim0_stride lo
  g1[6] = 0;
  g1[7] = 0;
  v4i z4 = {0, 0, 0, 0};
  v8i z8 = {0, 0, 0, 0, 0, 0, 0, 0};
  __builtin_amdgcn_tensor_load_to_lds(g0, g1, z4, z4, z8, 0);
}
#endif

// ---------------------------------------------------------------------------
// Elementwise f32 -> f16
// ---------------------------------------------------------------------------
__global__ void cvt_f32_f16(const float* __restrict__ s, _Float16* __restrict__ d,
                            int n) {
  int i = blockIdx.x * blockDim.x + threadIdx.x;
  if (i < n) d[i] = (_Float16)s[i];
}

// ---------------------------------------------------------------------------
// Tiled WMMA GEMM: C(f32, MxN) = A(f16, MxK row-major) * B(f16, KxN row-major)
// Block 256 threads = 8 waves (4m x 2n), block tile 128x128x32, wave tile
// 32x64 (2x4 WMMA tiles -> 8 v_wmma per iteration). M%128==N%128==K%32==0.
// A tile staged via TDM (wave 0) with hardware LDS padding; B tile staged
// transposed ([n][k]) by all threads.
// ---------------------------------------------------------------------------
#define GBM 128
#define GBN 128
#define GBK 32
#define LDSA 40
#define LDSB 40

__global__ void __launch_bounds__(256)
gemm_f16f32(const _Float16* __restrict__ A, const _Float16* __restrict__ B,
            float* __restrict__ C, int M, int N, int K) {
  __shared__ __align__(16) _Float16 As[GBM * LDSA];  // [m][k], 80B row stride
  __shared__ __align__(16) _Float16 Bs[GBN * LDSB];  // [n][k] (transposed)
  const int tid = threadIdx.x;
  const int bm = blockIdx.y * GBM;
  const int bn = blockIdx.x * GBN;
  const int w  = tid >> 5;
  const int l  = tid & 31;
  const int lm = l & 15;
  const int lh = l >> 4;
  const int koffA = lh ? 8 : 0;
  const int wm = (w >> 1) * 32;       // 4 wave rows
  const int wn = (w & 1) * 64;        // 2 wave cols
  const int brow = tid >> 3;          // 0..31  (B k-row)
  const int bcol = (tid & 7) * 16;    // 0..112 (B n-col)
#if !HAVE_TDM
  const int arow = tid >> 1;          // 0..127
  const int acol = (tid & 1) * 16;    // 0 / 16
#endif

  v8f acc[2][4] = {};

  for (int k0 = 0; k0 < K; k0 += GBK) {
    // B tile global loads (32B per thread)
    v8h b0 = *(const v8h*)(B + (size_t)(k0 + brow) * N + bn + bcol);
    v8h b1 = *(const v8h*)(B + (size_t)(k0 + brow) * N + bn + bcol + 8);
#if !HAVE_TDM
    v8h a0 = *(const v8h*)(A + (size_t)(bm + arow) * K + k0 + acol);
    v8h a1 = *(const v8h*)(A + (size_t)(bm + arow) * K + k0 + acol + 8);
#endif
    if (k0 + GBK < K)  // prefetch next B tile (global_prefetch_b8)
      __builtin_prefetch(B + (size_t)(k0 + GBK + brow) * N + bn + bcol, 0, 3);
    __syncthreads();   // previous iteration's LDS reads done
#if HAVE_TDM
    if (tid < 32) {    // wave 0 drives the Tensor Data Mover for the A tile
      tdm_load_tile_2d((unsigned)(uintptr_t)&As[0],
                       A + (size_t)bm * K + k0,
                       (unsigned)K, (unsigned)M, (unsigned)K,
                       GBK, GBM);
      __builtin_amdgcn_s_wait_tensorcnt(0);
    }
#else
    *(v8h*)(&As[arow * LDSA + acol])     = a0;
    *(v8h*)(&As[arow * LDSA + acol + 8]) = a1;
#endif
#pragma unroll
    for (int j = 0; j < 8; ++j) {
      Bs[(bcol + j) * LDSB + brow]     = b0[j];
      Bs[(bcol + 8 + j) * LDSB + brow] = b1[j];
    }
    __syncthreads();
    v16h af[2], bf[4];
#pragma unroll
    for (int i = 0; i < 2; ++i)
      af[i] = load_afrag(&As[(wm + i * 16 + lm) * LDSA], koffA);
#pragma unroll
    for (int j = 0; j < 4; ++j)
      bf[j] = load_bfrag(&Bs[(wn + j * 16 + lm) * LDSB + lh * 16]);
#pragma unroll
    for (int i = 0; i < 2; ++i)
#pragma unroll
      for (int j = 0; j < 4; ++j)
        acc[i][j] = wmma32(af[i], bf[j], acc[i][j]);
  }
  // C/D layout: lane<16 -> M=r, N=lane; lane>=16 -> M=8+r, N=lane-16
#pragma unroll
  for (int i = 0; i < 2; ++i)
#pragma unroll
    for (int j = 0; j < 4; ++j)
#pragma unroll
      for (int r = 0; r < 8; ++r)
        C[(size_t)(bm + wm + i * 16 + lh * 8 + r) * N + bn + wn + j * 16 + lm] =
            acc[i][j][r];
}

// ---------------------------------------------------------------------------
// RoPE + repack to head-major f16: src (S, H*HD) f32 -> dst [h][s][d] f16
// ---------------------------------------------------------------------------
__global__ void rope_pack(const float* __restrict__ src,
                          const float* __restrict__ freq,
                          _Float16* __restrict__ dst, int H, int ld, int n) {
  int idx = blockIdx.x * blockDim.x + threadIdx.x;
  if (idx >= n) return;
  int d2 = idx & 31;             // pair index 0..31
  int h  = (idx >> 5) % H;
  int s  = idx / (32 * H);
  float ang = freq[s * 32 + d2];
  float cs = cosf(ang), sn = sinf(ang);
  float xe = src[(size_t)s * ld + h * HD + 2 * d2];
  float xo = src[(size_t)s * ld + h * HD + 2 * d2 + 1];
  size_t o = ((size_t)h * S + s) * HD + 2 * d2;
  dst[o]     = (_Float16)(xe * cs - xo * sn);
  dst[o + 1] = (_Float16)(xe * sn + xo * cs);
}

// ---------------------------------------------------------------------------
// V transpose: vf (S, HKV*HD) f32 -> vt [h][d][s] f16  (contiguous in s)
// ---------------------------------------------------------------------------
__global__ void v_transpose(const float* __restrict__ vf,
                            _Float16* __restrict__ vt, int n) {
  int idx = blockIdx.x * blockDim.x + threadIdx.x;
  if (idx >= n) return;
  int s = idx & (S - 1);
  int d = (idx >> 11) & (HD - 1);
  int h = idx >> 17;
  vt[((size_t)h * HD + d) * S + s] =
      (_Float16)vf[(size_t)s * (HKV * HD) + h * HD + d];
}

// ---------------------------------------------------------------------------
// Flash attention: one wave per (q-head, 16-query tile). 32-key steps,
// online softmax, causal mask analytic. Writes attn16 (S, HQ*HD) f16.
// ---------------------------------------------------------------------------
__global__ void __launch_bounds__(32)
flash_attn(const _Float16* __restrict__ q16, const _Float16* __restrict__ k16,
           const _Float16* __restrict__ vt16, _Float16* __restrict__ attn16) {
  __shared__ __align__(16) _Float16 Plds[16 * 40];  // P tile, C->A relayout
  const int qb = blockIdx.x & (S / 16 - 1);
  const int hq = blockIdx.x / (S / 16);
  const int hk = hq / G;
  const int q0 = qb * 16;
  const int l  = threadIdx.x;
  const int lm = l & 15;
  const int lh = l >> 4;
  const int koffA = lh ? 8 : 0;

  // Q fragments (A-matrix): rows q0..q0+15, dims split 0..31 / 32..63
  const _Float16* qrow = q16 + ((size_t)hq * S + q0 + lm) * HD;
  v16h aq0 = load_afrag(qrow, koffA);
  v16h aq1 = load_afrag(qrow + 32, koffA);

  v8f o[4] = {};              // O: 16 queries x 64 dims (4 N-tiles)
  float mx[8], sm[8];         // per-row running max / sum (rows lh*8+r)
#pragma unroll
  for (int r = 0; r < 8; ++r) { mx[r] = -1e30f; sm[r] = 0.f; }

  const int nkb = (q0 + 16 + 31) >> 5;  // key blocks of 32 (causal bound)
  for (int kb = 0; kb < nkb; ++kb) {
    const int kbase = kb * 32;
    // --- S = Q K^T for two 16-key sub-tiles ---
    v8f sc[2];
#pragma unroll
    for (int t = 0; t < 2; ++t) {
      const int key = kbase + t * 16 + lm;          // always < S (see nkb)
      const _Float16* kptr = k16 + ((size_t)hk * S + key) * HD;
      v16h bk0 = load_bfrag(kptr + lh * 16);        // dims 0..31
      v16h bk1 = load_bfrag(kptr + 32 + lh * 16);   // dims 32..63
      v8f c = {};
      c = wmma32(aq0, bk0, c);
      c = wmma32(aq1, bk1, c);
      sc[t] = c;
    }
    // --- online softmax over this 32-key block ---
    float p0[8], p1[8];
#pragma unroll
    for (int r = 0; r < 8; ++r) {
      const int qi = q0 + lh * 8 + r;
      float v0 = (kbase + lm      <= qi) ? sc[0][r] * 0.125f : -1e30f;
      float v1 = (kbase + 16 + lm <= qi) ? sc[1][r] * 0.125f : -1e30f;
      float rmax = fmaxf(v0, v1);
      rmax = fmaxf(rmax, __shfl_xor(rmax, 1, 32));
      rmax = fmaxf(rmax, __shfl_xor(rmax, 2, 32));
      rmax = fmaxf(rmax, __shfl_xor(rmax, 4, 32));
      rmax = fmaxf(rmax, __shfl_xor(rmax, 8, 32));
      float mnew  = fmaxf(mx[r], rmax);
      float alpha = __expf(mx[r] - mnew);
      float e0 = __expf(v0 - mnew);
      float e1 = __expf(v1 - mnew);
      float rsum = e0 + e1;
      rsum += __shfl_xor(rsum, 1, 32);
      rsum += __shfl_xor(rsum, 2, 32);
      rsum += __shfl_xor(rsum, 4, 32);
      rsum += __shfl_xor(rsum, 8, 32);
      sm[r] = sm[r] * alpha + rsum;
      mx[r] = mnew;
      p0[r] = e0; p1[r] = e1;
#pragma unroll
      for (int c4 = 0; c4 < 4; ++c4) o[c4][r] *= alpha;
    }
    // --- relayout P (C-layout) -> A-layout via LDS ---
#pragma unroll
    for (int r = 0; r < 8; ++r) {
      const int m = lh * 8 + r;
      Plds[m * 40 + lm]      = (_Float16)p0[r];
      Plds[m * 40 + 16 + lm] = (_Float16)p1[r];
    }
    __syncthreads();
    v16h ap = load_afrag(&Plds[lm * 40], koffA);    // P: 16 x 32 keys
    // --- O += P * V (V^T layout gives contiguous B-fragments) ---
#pragma unroll
    for (int c4 = 0; c4 < 4; ++c4) {
      const _Float16* vptr =
          vt16 + ((size_t)hk * HD + c4 * 16 + lm) * S + kbase + lh * 16;
      v16h bvf = load_bfrag(vptr);
      o[c4] = wmma32(ap, bvf, o[c4]);
    }
    __syncthreads();
  }
  // --- normalize and store (S, HQ*HD) f16 for the output projection ---
#pragma unroll
  for (int r = 0; r < 8; ++r) {
    const float inv = 1.0f / sm[r];
    const size_t ro = (size_t)(q0 + lh * 8 + r) * D + hq * HD;
#pragma unroll
    for (int c4 = 0; c4 < 4; ++c4)
      attn16[ro + c4 * 16 + lm] = (_Float16)(o[c4][r] * inv);
  }
}

// ---------------------------------------------------------------------------
extern "C" void kernel_launch(void* const* d_in, const int* in_sizes, int n_in,
                              void* d_out, int out_size, void* d_ws,
                              size_t ws_size, hipStream_t stream) {
  (void)in_sizes; (void)n_in; (void)out_size; (void)ws_size;
  const float* x    = (const float*)d_in[0];
  const float* freq = (const float*)d_in[1];
  // d_in[2] = causal mask, implemented analytically in flash_attn
  const float* wq = (const float*)d_in[3];
  const float* wk = (const float*)d_in[4];
  const float* wv = (const float*)d_in[5];
  const float* wo = (const float*)d_in[6];
  float* out = (float*)d_out;

  char* p = (char*)d_ws;
  auto alloc = [&](size_t bytes) { char* r = p; p += bytes; return r; };
  _Float16* x16    = (_Float16*)alloc((size_t)S * D * 2);
  _Float16* wq16   = (_Float16*)alloc((size_t)D * D * 2);
  _Float16* wk16   = (_Float16*)alloc((size_t)D * HKV * HD * 2);
  _Float16* wv16   = (_Float16*)alloc((size_t)D * HKV * HD * 2);
  _Float16* wo16   = (_Float16*)alloc((size_t)D * D * 2);
  float*    qf     = (float*)alloc((size_t)S * D * 4);
  float*    kf     = (float*)alloc((size_t)S * HKV * HD * 4);
  float*    vf     = (float*)alloc((size_t)S * HKV * HD * 4);
  _Float16* q16r   = (_Float16*)alloc((size_t)HQ * S * HD * 2);
  _Float16* k16r   = (_Float16*)alloc((size_t)HKV * S * HD * 2);
  _Float16* vt16   = (_Float16*)alloc((size_t)HKV * HD * S * 2);
  _Float16* attn16 = (_Float16*)alloc((size_t)S * D * 2);

  auto cvt = [&](const float* src, _Float16* dst, int n) {
    cvt_f32_f16<<<(n + 255) / 256, 256, 0, stream>>>(src, dst, n);
  };
  cvt(x,  x16,  S * D);
  cvt(wq, wq16, D * D);
  cvt(wk, wk16, D * HKV * HD);
  cvt(wv, wv16, D * HKV * HD);
  cvt(wo, wo16, D * D);

  // Projections
  gemm_f16f32<<<dim3(D / GBN, S / GBM), 256, 0, stream>>>(x16, wq16, qf, S, D, D);
  gemm_f16f32<<<dim3((HKV * HD) / GBN, S / GBM), 256, 0, stream>>>(
      x16, wk16, kf, S, HKV * HD, D);
  gemm_f16f32<<<dim3((HKV * HD) / GBN, S / GBM), 256, 0, stream>>>(
      x16, wv16, vf, S, HKV * HD, D);

  // RoPE + layout repack
  int nq = S * HQ * 32;
  rope_pack<<<(nq + 255) / 256, 256, 0, stream>>>(qf, freq, q16r, HQ, HQ * HD, nq);
  int nk = S * HKV * 32;
  rope_pack<<<(nk + 255) / 256, 256, 0, stream>>>(kf, freq, k16r, HKV, HKV * HD, nk);
  int nv = HKV * HD * S;
  v_transpose<<<(nv + 255) / 256, 256, 0, stream>>>(vf, vt16, nv);

  // Attention
  flash_attn<<<HQ * (S / 16), 32, 0, stream>>>(q16r, k16r, vt16, attn16);

  // Output projection -> f32 out
  gemm_f16f32<<<dim3(D / GBN, S / GBM), 256, 0, stream>>>(attn16, wo16, out, S, D, D);
}
